// BiLSTM_CRF_81509889343881
// MI455X (gfx1250) — matrix operand order repaired and checked by reading, not compiled, not run on hardware
//
#include <hip/hip_runtime.h>

// ---------------------------------------------------------------------------
// BiLSTM-CRF forward NLL for MI455X (gfx1250, wave32, WMMA f32<=f16xf16)
//
// Pipeline (all on `stream`):
//  1) k_prep_wp   : pack [Wih;Whh] (both dirs) into WMMA B-fragment layout (f16)
//  2) k_prep_misc : pack fc_W into B-fragments (N padded 17->32), combine
//                   biases, zero per-step barrier counters (graph-replay safe)
//  3) k_embed     : embedding gather (+row-0 zero) packed into WMMA A-fragments
//  4) k_bilstm    : 8 persistent WGs (4 per direction). Weights staged once in
//                   256 KB of LDS (CDNA5: 320 KB/WGP). Each step: fused
//                   [x_t,h] @ W^T via v_wmma_f32_16x16x32_f16 (B frags from
//                   ds_load_b128), gate nonlinearities, cell state in VGPRs,
//                   h written in A-fragment layout; inter-WG step barrier.
//  5) k_fc        : logits = h_cat @ fc_W^T + b via WMMA (K=512, N=32 padded)
//  6) k_crf       : per-batch CRF forward algorithm (lane = tag), numerator,
//                   logsumexp partition (v_exp/v_log), per-batch NLL
//  7) k_reduce    : deterministic sum of 64 per-batch NLLs -> d_out[0]
// ---------------------------------------------------------------------------

typedef __attribute__((ext_vector_type(16))) _Float16 v16h;
typedef __attribute__((ext_vector_type(8)))  float    v8f;

#define WMMA_F16(a,b,c) __builtin_amdgcn_wmma_f32_16x16x32_f16( \
    false, (a), false, (b), (short)0, (c), false, false)

static constexpr int Bn = 64, Sn = 512, En = 256, Hn = 256, Tn = 17;

// workspace byte offsets (all 256-aligned)
static constexpr size_t OFF_WP  = 0x000000;  // 2*64*16*32*16 f16 = 2 MiB
static constexpr size_t OFF_FCW = 0x200000;  // 2*16*32*16 f16 = 32 KiB
static constexpr size_t OFF_BC  = 0x208000;  // 2048 f32 combined biases
static constexpr size_t OFF_BAR = 0x20A000;  // 1024 u32 step-barrier counters
static constexpr size_t OFF_CRF = 0x20B000;  // 64 f32 per-batch NLL
static constexpr size_t OFF_EP  = 0x300000;  // 512*4*8*32*16 f16 = 16 MiB
static constexpr size_t OFF_HS  = 0x1300000; // 512*4*16*32*16 f16 = 32 MiB
static constexpr size_t OFF_LG  = 0x3300000; // 32768*32 f32 = 4 MiB
// total ~55 MiB

// ---------------------------------------------------------------------------
// 1) pack concatenated recurrent weights [Wih | Whh] (K=512) into B-fragments:
//    wp[(((d*64+nt)*16+kt)*32+lane)*16+e], j = nt*16 + lane%16,
//    k = kt*32 + (e&7) + (e>=8 ? 16 : 0) + (lane/16)*8   (B 16-bit layout, table order)
__global__ void k_prep_wp(const float* __restrict__ Wih_f, const float* __restrict__ Whh_f,
                          const float* __restrict__ Wih_b, const float* __restrict__ Whh_b,
                          _Float16* __restrict__ wp) {
  int flat = blockIdx.x * 256 + threadIdx.x;            // < 1048576
  int e = flat & 15, lane = (flat >> 4) & 31, kt = (flat >> 9) & 15;
  int nt = (flat >> 13) & 63, d = flat >> 19;
  int half = lane >> 4, n = lane & 15;
  int j = nt * 16 + n;
  int k = kt * 32 + (e & 7) + ((e >> 3) << 4) + half * 8;
  const float* Wih = d ? Wih_b : Wih_f;
  const float* Whh = d ? Whh_b : Whh_f;
  float v = (k < En) ? Wih[j * En + k] : Whh[j * Hn + (k - En)];
  wp[flat] = (_Float16)v;
}

// 2) fc_W B-fragments (j>=17 zero-padded), combined biases, barrier zeroing
__global__ void k_prep_misc(const float* __restrict__ fcW,
                            const float* __restrict__ bihf, const float* __restrict__ bhhf,
                            const float* __restrict__ bihb, const float* __restrict__ bhhb,
                            _Float16* __restrict__ fcw, float* __restrict__ bc,
                            unsigned* __restrict__ bar) {
  int flat = blockIdx.x * 256 + threadIdx.x;
  if (flat < 16384) {
    int e = flat & 15, lane = (flat >> 4) & 31, kt = (flat >> 9) & 15, nt = flat >> 13;
    int half = lane >> 4, n = lane & 15;
    int j = nt * 16 + n, k = kt * 32 + (e & 7) + ((e >> 3) << 4) + half * 8;
    float v = (j < Tn) ? fcW[j * (2 * Hn) + k] : 0.0f;
    fcw[flat] = (_Float16)v;
  } else if (flat < 16384 + 2048) {
    int idx = flat - 16384;
    int d = idx >> 10, j = idx & 1023;
    bc[idx] = d ? (bihb[j] + bhhb[j]) : (bihf[j] + bhhf[j]);
  } else if (flat < 16384 + 2048 + 1024) {
    bar[flat - 16384 - 2048] = 0u;
  }
}

// 3) embedding gather -> A-fragment layout (16-bit A 16x32):
//    k32 = (e&7) + (e>=8 ? 16 : 0) + (lane/16)*8
__global__ void k_embed(const int* __restrict__ x, const float* __restrict__ emb,
                        _Float16* __restrict__ ep) {
  int flat = blockIdx.x * 256 + threadIdx.x;            // < 8388608
  int e = flat & 15, lane = (flat >> 4) & 31, kt = (flat >> 9) & 7;
  int m = (flat >> 12) & 3, s = flat >> 14;
  int half = lane >> 4;
  int b = m * 16 + (lane & 15);
  int k32 = (e & 7) + ((e >> 3) << 4) + half * 8;
  int kcol = kt * 32 + k32;
  int tok = x[b * Sn + s];
  float v = (tok == 0) ? 0.0f : emb[(size_t)tok * En + kcol];
  ep[flat] = (_Float16)v;
}

// 4) persistent BiLSTM recurrence. 8 WGs x 512 thr: blockIdx/4 = dir,
//    blockIdx%4 = hidden quarter. Each wave owns (m-tile, hid-tile) fixed for
//    all 512 steps; cell state c stays in VGPRs; weight B-fragments staged in
//    256 KB dynamic LDS; h exchanged via L2 in A-fragment layout (doubles as
//    FC input history).
__global__ void __launch_bounds__(512) k_bilstm(const _Float16* __restrict__ ep,
                                                const _Float16* __restrict__ wp,
                                                const float* __restrict__ bc,
                                                _Float16* __restrict__ hs,
                                                unsigned* __restrict__ bar) {
  extern __shared__ _Float16 wlds[];   // 16 lnt * 16 kt * 32 lane * 16 e = 256 KB
  const int tid = threadIdx.x, lane = tid & 31, wv = tid >> 5;
  const int d = blockIdx.x >> 2, q = blockIdx.x & 3;
  const int m = wv >> 2, htl = wv & 3, ht = q * 4 + htl;
  const int half = lane >> 4, n = lane & 15;

  // cooperative preload of this WG's weight fragments into LDS (coalesced)
#pragma unroll
  for (int cc = 0; cc < 16; ++cc) {
    int c = cc * 512 + tid;                       // v16h chunk id < 8192
    int lane_c = c & 31, kt_c = (c >> 5) & 15, lnt_c = c >> 9;
    int nt_c = (lnt_c >> 2) * 16 + q * 4 + (lnt_c & 3);
    const v16h* src = (const v16h*)(wp +
        (((size_t)(d * 64 + nt_c) * 16 + kt_c) * 32 + lane_c) * 16);
    *(v16h*)(wlds + (size_t)c * 16) = *src;
  }
  __syncthreads();

  // per-wave LDS base pointers for the 4 gate tiles (stride 512 f16 per kt)
  const _Float16* wl[4];
  float bias[4];
#pragma unroll
  for (int g = 0; g < 4; ++g) {
    wl[g]   = wlds + (size_t)(g * 4 + htl) * 8192 + (size_t)lane * 16;
    bias[g] = bc[d * 1024 + (g * 16 + ht) * 16 + n];
  }
  // store-position constants for h in A-fragment layout
  const int kpos = (ht & 1) * 16 + n;
  const int h16p = (kpos >> 3) & 1;
  const int epos = (kpos & 7) + (kpos >= 16 ? 8 : 0);
  const int kt16 = d * 8 + (ht >> 1);

  v8f c;
#pragma unroll
  for (int r = 0; r < 8; ++r) c[r] = 0.0f;

  for (int st = 0; st < Sn; ++st) {
    const int t = d ? (Sn - 1 - st) : st;

    // keep all loads inside the loop (defeat LICM hoist -> scratch spill)
    asm volatile("" ::: "memory");

    v8f acc[4];
#pragma unroll
    for (int g = 0; g < 4; ++g) {
#pragma unroll
      for (int r = 0; r < 8; ++r) acc[g][r] = bias[g];
    }

    // input projection part: K = 0..255 from packed embeddings
    const v16h* Abase = (const v16h*)(ep + (size_t)(t * 4 + m) * 8 * 512);
    __builtin_prefetch(Abase, 0, 3);
#pragma unroll
    for (int kt = 0; kt < 8; ++kt) {
      v16h a = Abase[kt * 32 + lane];
#pragma unroll
      for (int g = 0; g < 4; ++g) {
        v16h bf = *(const v16h*)(wl[g] + kt * 512);
        acc[g] = WMMA_F16(a, bf, acc[g]);
      }
    }
    // recurrent part: K = 256..511 from previous h (skip at first step, h0=0)
    if (st > 0) {
      const int tp = d ? (t + 1) : (t - 1);
      const v16h* Hbase = (const v16h*)(hs + ((size_t)(tp * 4 + m) * 16 + d * 8) * 512);
#pragma unroll
      for (int kt = 0; kt < 8; ++kt) {
        v16h a = Hbase[kt * 32 + lane];
#pragma unroll
        for (int g = 0; g < 4; ++g) {
          v16h bf = *(const v16h*)(wl[g] + (8 + kt) * 512);
          acc[g] = WMMA_F16(a, bf, acc[g]);
        }
      }
    }

    // gates: c = sig(f)*c + sig(i)*tanh(g); h = sig(o)*tanh(c)
    _Float16* hout = hs + ((size_t)(t * 4 + m) * 16 + kt16) * 512;
#pragma unroll
    for (int r = 0; r < 8; ++r) {
      float ig = 1.0f / (1.0f + __expf(-acc[0][r]));
      float fg = 1.0f / (1.0f + __expf(-acc[1][r]));
      float gg = tanhf(acc[2][r]);
      float og = 1.0f / (1.0f + __expf(-acc[3][r]));
      float cn = fg * c[r] + ig * gg;
      c[r] = cn;
      float hv = og * tanhf(cn);
      hout[(h16p * 16 + r + 8 * half) * 16 + epos] = (_Float16)hv;
    }

    // inter-WG step barrier (4 WGs per direction), counters pre-zeroed
    __threadfence();
    __syncthreads();
    if (tid == 0) {
      unsigned* p = bar + d * 512 + st;
      __hip_atomic_fetch_add(p, 1u, __ATOMIC_RELEASE, __HIP_MEMORY_SCOPE_AGENT);
      while (__hip_atomic_load(p, __ATOMIC_ACQUIRE, __HIP_MEMORY_SCOPE_AGENT) < 4u)
        __builtin_amdgcn_s_sleep(1);
    }
    __syncthreads();
    __threadfence();
  }
}

// 5) logits = h_cat @ fc_W^T + fc_b  (M=32768 rows ordered r = s*64+b,
//    K=512, N=32 with cols 17..31 zero). One wave per 16-row tile.
__global__ void k_fc(const _Float16* __restrict__ hs, const _Float16* __restrict__ fcw,
                     const float* __restrict__ fcb, float* __restrict__ lg) {
  int tid = threadIdx.x, lane = tid & 31;
  int wvg = blockIdx.x * 8 + (tid >> 5);                // m-tile, 0..2047
  int s = wvg >> 2, m = wvg & 3;
  int half = lane >> 4, n = lane & 15;

  v8f acc[2];
#pragma unroll
  for (int nt = 0; nt < 2; ++nt) {
    int j = nt * 16 + n;
    float bv = (j < Tn) ? fcb[j] : 0.0f;
#pragma unroll
    for (int r = 0; r < 8; ++r) acc[nt][r] = bv;
  }

  const v16h* Ab = (const v16h*)(hs + (size_t)(s * 4 + m) * 16 * 512);
#pragma unroll
  for (int kt = 0; kt < 16; ++kt) {
    v16h a = Ab[kt * 32 + lane];
#pragma unroll
    for (int nt = 0; nt < 2; ++nt) {
      v16h bf = *(const v16h*)(fcw + ((size_t)(nt * 16 + kt) * 32 + lane) * 16);
      acc[nt] = WMMA_F16(a, bf, acc[nt]);
    }
  }
#pragma unroll
  for (int nt = 0; nt < 2; ++nt) {
#pragma unroll
    for (int r = 0; r < 8; ++r) {
      int b = m * 16 + r + 8 * half;
      lg[(size_t)(s * 64 + b) * 32 + nt * 16 + n] = acc[nt][r];
    }
  }
}

// 6) CRF: one wave per batch; lane = tag (0..16 active). Partition via
//    forward logsumexp recursion; numerator lane-parallel over time.
__global__ void k_crf(const float* __restrict__ lg, const int* __restrict__ tags,
                      const int* __restrict__ mask, const float* __restrict__ start_t,
                      const float* __restrict__ end_t, const float* __restrict__ trans,
                      float* __restrict__ outb) {
  __shared__ float tr[17 * 32];
  int b = blockIdx.x, lane = threadIdx.x;
  for (int i = lane; i < 17 * 32; i += 32) {
    int r = i >> 5, cc = i & 31;
    tr[i] = (cc < Tn) ? trans[r * Tn + cc] : 0.0f;
  }
  __syncthreads();

  // numerator
  float num = 0.0f; int msum = 0;
  for (int t = lane; t < Sn; t += 32) {
    int tg = tags[b * Sn + t];
    if (t == 0) {
      num += start_t[tg] + lg[(size_t)b * 32 + tg];
    } else {
      int tgp = tags[b * Sn + t - 1];
      float mt = (float)mask[b * Sn + t];
      num += mt * (trans[tgp * Tn + tg] + lg[((size_t)t * 64 + b) * 32 + tg]);
    }
    msum += mask[b * Sn + t] ? 1 : 0;
  }
#pragma unroll
  for (int o = 16; o >= 1; o >>= 1) {
    num  += __shfl_xor(num, o, 32);
    msum += __shfl_xor(msum, o, 32);
  }

  // partition (forward algorithm)
  float score = (lane < Tn) ? (start_t[lane] + lg[(size_t)b * 32 + lane]) : -1e30f;
  for (int t = 1; t < Sn; ++t) {
    float em = lg[((size_t)t * 64 + b) * 32 + lane];
    int mt = mask[b * Sn + t];
    float v[17];
#pragma unroll
    for (int i = 0; i < 17; ++i) v[i] = __shfl(score, i, 32) + tr[i * 32 + lane];
    float mx = v[0];
#pragma unroll
    for (int i = 1; i < 17; ++i) mx = fmaxf(mx, v[i]);
    float sm = 0.0f;
#pragma unroll
    for (int i = 0; i < 17; ++i) sm += __expf(v[i] - mx);
    float nxt = mx + __logf(sm) + em;
    score = mt ? nxt : score;
  }
  float fv = (lane < Tn) ? (score + end_t[lane]) : -1e30f;
  float mx = fv;
#pragma unroll
  for (int o = 16; o >= 1; o >>= 1) mx = fmaxf(mx, __shfl_xor(mx, o, 32));
  float se = (lane < Tn) ? __expf(fv - mx) : 0.0f;
#pragma unroll
  for (int o = 16; o >= 1; o >>= 1) se += __shfl_xor(se, o, 32);
  float denom = mx + __logf(se);

  if (lane == 0) {
    int last = msum - 1;
    num += end_t[tags[b * Sn + last]];
    outb[b] = denom - num;
  }
}

// 7) deterministic final sum -> d_out[0]
__global__ void k_reduce(const float* __restrict__ outb, float* __restrict__ out) {
  __shared__ float sd[64];
  sd[threadIdx.x] = outb[threadIdx.x];
  __syncthreads();
  if (threadIdx.x == 0) {
    float s = 0.0f;
    for (int i = 0; i < 64; ++i) s += sd[i];
    out[0] = s;
  }
}

extern "C" void kernel_launch(void* const* d_in, const int* in_sizes, int n_in,
                              void* d_out, int out_size, void* d_ws, size_t ws_size,
                              hipStream_t stream) {
  const int*   x       = (const int*)  d_in[0];
  const int*   tags    = (const int*)  d_in[1];
  const int*   mask    = (const int*)  d_in[2];
  const float* emb     = (const float*)d_in[3];
  const float* Wih_f   = (const float*)d_in[4];
  const float* Whh_f   = (const float*)d_in[5];
  const float* b_ih_f  = (const float*)d_in[6];
  const float* b_hh_f  = (const float*)d_in[7];
  const float* Wih_b   = (const float*)d_in[8];
  const float* Whh_b   = (const float*)d_in[9];
  const float* b_ih_b  = (const float*)d_in[10];
  const float* b_hh_b  = (const float*)d_in[11];
  const float* fc_W    = (const float*)d_in[12];
  const float* fc_b    = (const float*)d_in[13];
  const float* start_t = (const float*)d_in[14];
  const float* end_t   = (const float*)d_in[15];
  const float* trans   = (const float*)d_in[16];

  char* ws = (char*)d_ws;
  _Float16* wp   = (_Float16*)(ws + OFF_WP);
  _Float16* fcw  = (_Float16*)(ws + OFF_FCW);
  float*    bc   = (float*)   (ws + OFF_BC);
  unsigned* bar  = (unsigned*)(ws + OFF_BAR);
  float*    crf  = (float*)   (ws + OFF_CRF);
  _Float16* ep   = (_Float16*)(ws + OFF_EP);
  _Float16* hsb  = (_Float16*)(ws + OFF_HS);
  float*    lg   = (float*)   (ws + OFF_LG);
  float*    out  = (float*)d_out;

  // allow 256 KB dynamic LDS for the recurrence kernel (CDNA5: 320 KB/WG)
  static const size_t kWLDS = 16u * 16u * 32u * 16u * sizeof(_Float16);  // 262144
  (void)hipFuncSetAttribute((const void*)k_bilstm,
                            hipFuncAttributeMaxDynamicSharedMemorySize,
                            (int)kWLDS);

  k_prep_wp  <<<4096, 256, 0, stream>>>(Wih_f, Whh_f, Wih_b, Whh_b, wp);
  k_prep_misc<<<76,   256, 0, stream>>>(fc_W, b_ih_f, b_hh_f, b_ih_b, b_hh_b, fcw, bc, bar);
  k_embed    <<<32768,256, 0, stream>>>(x, emb, ep);
  k_bilstm   <<<8,    512, kWLDS, stream>>>(ep, wp, bc, hsb, bar);
  k_fc       <<<256,  256, 0, stream>>>(hsb, fcw, fc_b, lg);
  k_crf      <<<64,   32,  0, stream>>>(lg, tags, mask, start_t, end_t, trans, crf);
  k_reduce   <<<1,    64,  0, stream>>>(crf, out);
}
